// RingDilatedAttentionTritonOptimized_68959994905211
// MI455X (gfx1250) — matrix operand order repaired and checked by reading, not compiled, not run on hardware
//
#include <hip/hip_runtime.h>
#include <math.h>

typedef __attribute__((ext_vector_type(16))) __bf16 v16bf;
typedef __attribute__((ext_vector_type(8)))  float  v8f;

#define Sseq 4096
#define NH   12
#define Dh   128
#define SRow (NH * Dh)      // 1536 floats between consecutive seq positions
#define KTILE 64
#define QTILE 128           // 8 waves * 16 query rows

__device__ __forceinline__ float red_max16(float v) {
  v = fmaxf(v, __shfl_xor(v, 1, 32));
  v = fmaxf(v, __shfl_xor(v, 2, 32));
  v = fmaxf(v, __shfl_xor(v, 4, 32));
  v = fmaxf(v, __shfl_xor(v, 8, 32));
  return v;
}
__device__ __forceinline__ float red_sum16(float v) {
  v += __shfl_xor(v, 1, 32);
  v += __shfl_xor(v, 2, 32);
  v += __shfl_xor(v, 4, 32);
  v += __shfl_xor(v, 8, 32);
  return v;
}

__global__ void zero_out_kernel(float4* __restrict__ out, int n4) {
  int i = blockIdx.x * blockDim.x + threadIdx.x;
  if (i < n4) out[i] = make_float4(0.f, 0.f, 0.f, 0.f);
}

__global__ __launch_bounds__(256, 1)
void dilated_flash_attn(const float* __restrict__ Q, const float* __restrict__ K,
                        const float* __restrict__ V, float* __restrict__ O) {
  // ---- decode work item -------------------------------------------------
  int bid = blockIdx.x;
  int b, h, qbase, qstride, kstart, kstride, kcount;
  if (bid < 256) {                       // group 0: full attention, heads 0-3
    int bh = bid >> 5, qt = bid & 31;
    b = bh >> 2; h = bh & 3;
    qstride = 1; qbase = qt * QTILE;
    kstart = 0; kstride = 1; kcount = 4096;
  } else if (bid < 384) {                // group 1: dil=2, heads 4-7
    int t = bid - 256;
    int sub = t >> 6, bh = (t >> 3) & 7, qt = t & 7;
    b = bh >> 2; h = 4 + (bh & 3);
    qstride = 2; kstride = 2; kcount = 2048;
    if (sub == 0) { qbase = 1 + qt * QTILE * 2;    kstart = 1; }   // seg 0, odd
    else          { qbase = 2048 + qt * QTILE * 2; kstart = 0; }   // seg 1, even
  } else {                               // group 2: dil=4, heads 8-11
    int t = bid - 384;
    int bh = t >> 3, qt = t & 7;
    b = bh >> 2; h = 8 + (bh & 3);
    qstride = 4; qbase = 2 + qt * QTILE * 4;
    kstart = 2; kstride = 4; kcount = 1024;
  }

  const int tid  = threadIdx.x;
  const int wave = tid >> 5;
  const int lane = tid & 31;
  const int lrow = lane & 15;
  const int lhi  = lane >> 4;            // 0: lanes 0-15, 1: lanes 16-31
  const int kA   = lhi ? 8 : 0;          // A-layout per-lane K offset
  const int kB   = lhi ? 16 : 0;         // unused; B loads use kA as well
  const float scale = 0.08838834764831845f;  // 1/sqrt(128), folded into Q

  const size_t bhOff = (size_t)b * Sseq * SRow + (size_t)h * Dh;
  const float* Qb = Q + bhOff;
  const float* Kb = K + bhOff;
  const float* Vb = V + bhOff;
  float*       Ob = O + bhOff;

  __shared__ __bf16 ldsK[KTILE][Dh + 4];       // 64 x 132
  __shared__ __bf16 ldsVt[Dh][KTILE + 4];      // transposed V: 128 x 68
  __shared__ __bf16 ldsP[8][16][KTILE];        // per-wave P staging

  // ---- load Q fragments (A layout, scale pre-folded) --------------------
  const int qs = qbase + (wave * 16 + lrow) * qstride;
  const float* qrow = Qb + (size_t)qs * SRow;
  v16bf aq[4];
#pragma unroll
  for (int c = 0; c < 4; ++c) {
#pragma unroll
    for (int e = 0; e < 16; ++e) {
      int d = 32 * c + ((e < 8) ? (kA + e) : (16 + kA + (e - 8)));
      aq[c][e] = (__bf16)(qrow[d] * scale);
    }
  }

  // ---- state ------------------------------------------------------------
  v8f acc[8];
#pragma unroll
  for (int f = 0; f < 8; ++f)
#pragma unroll
    for (int j = 0; j < 8; ++j) acc[f][j] = 0.f;
  float mj[8], lj[8];
#pragma unroll
  for (int j = 0; j < 8; ++j) { mj[j] = -INFINITY; lj[j] = 0.f; }

  // ---- key loop (64 keys/iter) ------------------------------------------
  const int nkt = kcount / KTILE;
  for (int kt = 0; kt < nkt; ++kt) {
    __syncthreads();
    {   // cooperative stage of K (row-major) and V (transposed) tiles
      int row = tid >> 2;                 // 0..63
      int d0  = (tid & 3) << 5;           // 0,32,64,96
      int ks  = kstart + (kt * KTILE + row) * kstride;
      const float* krow = Kb + (size_t)ks * SRow + d0;
      const float* vrow = Vb + (size_t)ks * SRow + d0;
#pragma unroll
      for (int e = 0; e < 32; ++e) {
        ldsK[row][d0 + e]  = (__bf16)krow[e];
        ldsVt[d0 + e][row] = (__bf16)vrow[e];
      }
    }
    __syncthreads();

    // ---- S = Q K^T : 16 queries x 64 keys (4 C fragments) ---------------
    v8f s[4];
#pragma unroll
    for (int n = 0; n < 4; ++n)
#pragma unroll
      for (int j = 0; j < 8; ++j) s[n][j] = 0.f;
#pragma unroll
    for (int c = 0; c < 4; ++c) {
#pragma unroll
      for (int n = 0; n < 4; ++n) {
        v16bf bk;
#pragma unroll
        for (int e = 0; e < 16; ++e)
          bk[e] = ldsK[16 * n + lrow][32 * c + ((e < 8) ? (kA + e) : (16 + kA + (e - 8)))];
        s[n] = __builtin_amdgcn_wmma_f32_16x16x32_bf16(false, aq[c], false, bk,
                                                       (short)0, s[n], false, false);
      }
    }

    // ---- online softmax (rows j / j+8 per lane half) --------------------
#pragma unroll
    for (int j = 0; j < 8; ++j) {
      float v0 = s[0][j], v1 = s[1][j], v2 = s[2][j], v3 = s[3][j];
      float mx   = red_max16(fmaxf(fmaxf(v0, v1), fmaxf(v2, v3)));
      float mnew = fmaxf(mj[j], mx);
      float corr = __expf(mj[j] - mnew);
      float p0 = __expf(v0 - mnew);
      float p1 = __expf(v1 - mnew);
      float p2 = __expf(v2 - mnew);
      float p3 = __expf(v3 - mnew);
      float rs = red_sum16((p0 + p1) + (p2 + p3));
      lj[j] = lj[j] * corr + rs;
      mj[j] = mnew;
#pragma unroll
      for (int f = 0; f < 8; ++f) acc[f][j] *= corr;
      int pr = j + (lhi << 3);
      ldsP[wave][pr][lrow]      = (__bf16)p0;
      ldsP[wave][pr][lrow + 16] = (__bf16)p1;
      ldsP[wave][pr][lrow + 32] = (__bf16)p2;
      ldsP[wave][pr][lrow + 48] = (__bf16)p3;
    }
    asm volatile("s_wait_dscnt 0" ::: "memory");  // wave-local P store->load

    // ---- P (C layout -> A layout via LDS): two 16x32 A fragments --------
    v16bf ap[2];
#pragma unroll
    for (int t = 0; t < 2; ++t)
#pragma unroll
      for (int e = 0; e < 16; ++e) {
        int kk = (e < 8) ? (kA + e) : (16 + kA + (e - 8));
        ap[t][e] = ldsP[wave][lrow][32 * t + kk];
      }

    // ---- O += P V : 8 chunks of 16 output dims --------------------------
#pragma unroll
    for (int f = 0; f < 8; ++f) {
#pragma unroll
      for (int t = 0; t < 2; ++t) {
        v16bf bv;
#pragma unroll
        for (int e = 0; e < 16; ++e)
          bv[e] = ldsVt[f * 16 + lrow][32 * t + ((e < 8) ? (kA + e) : (16 + kA + (e - 8)))];
        acc[f] = __builtin_amdgcn_wmma_f32_16x16x32_bf16(false, ap[t], false, bv,
                                                         (short)0, acc[f], false, false);
      }
    }
  }

  // ---- epilogue: normalize and scatter ---------------------------------
  float inv[8];
#pragma unroll
  for (int j = 0; j < 8; ++j) inv[j] = 1.0f / lj[j];
#pragma unroll
  for (int j = 0; j < 8; ++j) {
    int row = j + (lhi << 3);
    int qso = qbase + (wave * 16 + row) * qstride;
    float* orow = Ob + (size_t)qso * SRow;
#pragma unroll
    for (int f = 0; f < 8; ++f)
      orow[f * 16 + lrow] = acc[f][j] * inv[j];
  }
}

extern "C" void kernel_launch(void* const* d_in, const int* in_sizes, int n_in,
                              void* d_out, int out_size, void* d_ws, size_t ws_size,
                              hipStream_t stream) {
  const float* q = (const float*)d_in[0];
  const float* k = (const float*)d_in[1];
  const float* v = (const float*)d_in[2];
  float* out = (float*)d_out;

  int n4 = out_size / 4;  // 12.58M floats -> 3.15M float4
  zero_out_kernel<<<(n4 + 255) / 256, 256, 0, stream>>>((float4*)out, n4);
  dilated_flash_attn<<<448, 256, 0, stream>>>(q, k, v, out);
}
